// FSQ_Mamba_26749056319764
// MI455X (gfx1250) — matrix operand order repaired and checked by reading, not compiled, not run on hardware
//
#include <hip/hip_runtime.h>
#include <hip/hip_bf16.h>
#include <cstdint>

typedef __bf16 bf16_t;
typedef __attribute__((ext_vector_type(16))) __bf16 v16bf;
typedef __attribute__((ext_vector_type(8)))  __bf16 v8bf;
typedef __attribute__((ext_vector_type(8)))  float  v8f;

#if defined(__has_builtin)
#if __has_builtin(__builtin_amdgcn_tensor_load_to_lds) && __has_builtin(__builtin_amdgcn_s_wait_tensorcnt)
#define HAVE_TDM 1
#endif
#endif
#ifndef HAVE_TDM
#define HAVE_TDM 0
#endif

#define RD 256
#define DI 256
#define NSV 16
#define DTR 16
#define KW 4
#define DEPTH 8
#define BATCH 2
#define LSP 3600
#define LSEQ 7200
#define ROWS 14400

#define BM 128
#define BN 128
#define BK 32
#define LDT 40   // padded LDS row: 80B pitch (64B data + 16B pad), 16B aligned

__device__ __forceinline__ bf16_t f2bf(float f) { return (bf16_t)f; }
__device__ __forceinline__ float siluf(float x) { return x / (1.f + __expf(-x)); }
__device__ __forceinline__ float softplusf(float x) {
    return (x > 20.f) ? x : log1pf(__expf(x));
}

// ---- WMMA fragment loaders (ISA 7.12.2 layouts) ----------------------------
// A (16x32 bf16, row-major MxK in LDS): lane l<16: row l, K 0..7 & 16..23;
// lane l+16: row l, K 8..15 & 24..31.
__device__ __forceinline__ v16bf ldfragA(const bf16_t* p) {
    v8bf c0 = *(const v8bf*)p;
    v8bf c1 = *(const v8bf*)(p + 16);
    v16bf r;
#pragma unroll
    for (int i = 0; i < 8; ++i) { r[i] = c0[i]; r[8 + i] = c1[i]; }
    return r;
}
// B (32x16 bf16) staged as B^T row-major [N][K]: lane n<16: col n, K 0..7 & 16..23;
// lane n+16: col n, K 8..15 & 24..31 (same k order as A).
__device__ __forceinline__ v16bf ldfragB(const bf16_t* p) {
    v8bf c0 = *(const v8bf*)p;
    v8bf c1 = *(const v8bf*)(p + 16);
    v16bf r;
#pragma unroll
    for (int i = 0; i < 8; ++i) { r[i] = c0[i]; r[8 + i] = c1[i]; }
    return r;
}

#if HAVE_TDM
typedef __attribute__((ext_vector_type(4))) unsigned int u32x4;
typedef __attribute__((ext_vector_type(8))) int i32x8;
typedef __attribute__((ext_vector_type(4))) int i32x4;

// Issue a TDM 2D tile load: rows x 32 bf16 elements from a [.. x K] tensor
// (row stride K elems) into LDS at lds_off, with 16B padding after each
// 64B row so rows land at LDT(=40)-element pitch.
__device__ __forceinline__ void tdm_load_b_tile(const bf16_t* g, unsigned lds_off,
                                                int rows, int K)
{
    unsigned long long ga = (unsigned long long)(uintptr_t)g;
    u32x4 g0;
    g0[0] = 1u;                                            // count=1, user mode
    g0[1] = lds_off;                                       // lds_addr (bytes)
    g0[2] = (unsigned)(ga & 0xffffffffu);                  // global_addr lo
    g0[3] = (unsigned)((ga >> 32) & 0x01ffffffu) | (2u << 30); // addr hi | type=2
    i32x8 g1;
    g1[0] = (int)((1u << 16) |     // data_size = 2 bytes
                  (1u << 20) |     // pad_enable
                  (3u << 22) |     // pad_interval: 16 DWORDs (64B) before pad
                  (3u << 25));     // pad_amount: 4 DWORDs (16B)
    g1[1] = (int)(((unsigned)K & 0xffffu) << 16);          // tensor_dim0 lo
    g1[2] = (int)((((unsigned)K >> 16) & 0xffffu) |
                  (((unsigned)rows & 0xffffu) << 16));     // dim0 hi | dim1 lo
    g1[3] = (int)(32u << 16);                              // dim1 hi | tile_dim0=32
    g1[4] = (int)((unsigned)rows & 0xffffu);               // tile_dim1 | tile_dim2=0
    g1[5] = (int)(unsigned)K;                              // dim0_stride lo
    g1[6] = 0;                                             // stride0 hi | stride1 lo
    g1[7] = 0;
    i32x4 z4 = {0, 0, 0, 0};
#if __clang_major__ >= 23
    i32x8 z8 = {0, 0, 0, 0, 0, 0, 0, 0};
    __builtin_amdgcn_tensor_load_to_lds(g0, g1, z4, z4, z8, 0);
#else
    __builtin_amdgcn_tensor_load_to_lds(g0, g1, z4, z4, 0);
#endif
}
#endif

// ---- weight repack: f32 -> bf16, into B^T [N][K] layout --------------------
__global__ __launch_bounds__(256) void pack_bt(
    const float* __restrict__ src, bf16_t* __restrict__ dst,
    int N, int K, long sn, long sk, int rot, long lss, long lsd)
{
    long layer = blockIdx.z;
    src += layer * lss;
    dst += layer * lsd;
    long idx = (long)blockIdx.x * 256 + threadIdx.x;
    if (idx >= (long)N * K) return;
    int n = (int)(idx / K), k = (int)(idx % K);
    int ks = rot ? (k + rot) % K : k;
    dst[(long)n * K + k] = f2bf(src[(long)n * sn + (long)ks * sk]);
}

// ---- generic WMMA GEMM: C[MxN] (+=) relu( A[MxK](strided f32) * BT^T ) -----
// 128x128 block tile, 8 wave32 waves; each wave = 32x64 (2x4 WMMA tiles).
__global__ __launch_bounds__(256) void gemm_bt(
    const float* __restrict__ A, const bf16_t* __restrict__ BT,
    float* __restrict__ C, int M, int N, int K,
    long sAm, long sAk, int ldc, int relu, int accum)
{
    __shared__ bf16_t As[BM * LDT];
    __shared__ bf16_t Bs[BN * LDT];

    const int t    = threadIdx.x;
    const int lane = t & 31;
    const int wid  = t >> 5;
    const int wm   = (wid & 3) * 32;   // 4 waves along M (32 rows each)
    const int wn   = (wid >> 2) * 64;  // 2 waves along N (64 cols each)
    const long m0  = (long)blockIdx.x * BM;
    const long n0  = (long)blockIdx.y * BN;

    v8f acc[2][4];
#pragma unroll
    for (int i = 0; i < 2; ++i)
#pragma unroll
        for (int j = 0; j < 4; ++j)
#pragma unroll
            for (int e = 0; e < 8; ++e) acc[i][j][e] = 0.f;

    const int am = t & 127;            // A row within tile
    const int ak = (t >> 7) * 16;      // A k-chunk (0 or 16)
#if !HAVE_TDM
    const int bn = t & 127;            // B row (n) within tile
    const int bk = (t >> 7) * 16;      // B k-chunk (0 or 16)
#endif
#if HAVE_TDM
    const unsigned bs_lds = (unsigned)(uintptr_t)(&Bs[0]);
    int brows = N - (int)n0;
    if (brows > BN) brows = BN;
#endif

    for (int k0 = 0; k0 < K; k0 += BK) {
#if HAVE_TDM
        // ---- stage B via Tensor Data Mover (wave 0 issues; padded pitch) ----
        if (wid == 0)
            tdm_load_b_tile(BT + n0 * K + k0, bs_lds, brows, K);
#endif
        // ---- stage A (convert f32 -> bf16) ----
        {
            float fv[16];
            const long gm = m0 + am;
            if (gm < M) {
                const float* ap = A + gm * sAm + (long)(k0 + ak) * sAk;
                if (sAk == 1) {
                    const float4* ap4 = (const float4*)ap;
#pragma unroll
                    for (int q = 0; q < 4; ++q) {
                        float4 f = ap4[q];
                        fv[q * 4 + 0] = f.x; fv[q * 4 + 1] = f.y;
                        fv[q * 4 + 2] = f.z; fv[q * 4 + 3] = f.w;
                    }
                } else {
#pragma unroll
                    for (int kk = 0; kk < 16; ++kk) fv[kk] = ap[(long)kk * sAk];
                }
            } else {
#pragma unroll
                for (int kk = 0; kk < 16; ++kk) fv[kk] = 0.f;
            }
            v8bf lo, hi;
#pragma unroll
            for (int kk = 0; kk < 8; ++kk) { lo[kk] = f2bf(fv[kk]); hi[kk] = f2bf(fv[8 + kk]); }
            *(v8bf*)&As[am * LDT + ak]     = lo;
            *(v8bf*)&As[am * LDT + ak + 8] = hi;
        }
#if !HAVE_TDM
        // ---- stage B synchronously (fallback) ----
        {
            const long gn = n0 + bn;
            v8bf bv0, bv1;
            if (gn < N) {
                bv0 = *(const v8bf*)(BT + gn * K + k0 + bk);
                bv1 = *(const v8bf*)(BT + gn * K + k0 + bk + 8);
            } else {
#pragma unroll
                for (int e = 0; e < 8; ++e) { bv0[e] = (bf16_t)0.f; bv1[e] = (bf16_t)0.f; }
            }
            *(v8bf*)&Bs[bn * LDT + bk]     = bv0;
            *(v8bf*)&Bs[bn * LDT + bk + 8] = bv1;
            if (k0 + BK < K && gn < N)
                __builtin_prefetch((const void*)(BT + gn * K + k0 + BK + bk), 0, 1);
        }
#endif
#if HAVE_TDM
        __builtin_amdgcn_s_wait_tensorcnt(0);  // issuing wave drains TDM
#endif
        __syncthreads();

        // ---- compute: 2x4 tiles of 16x16, one K=32 WMMA each (8 WMMAs) ----
        {
            const int ar = lane & 15;
            const int ac = (lane >> 4) * 8;
            const int bc = (lane >> 4) * 8;
            v16bf af[2], bfg[4];
#pragma unroll
            for (int i = 0; i < 2; ++i)
                af[i] = ldfragA(&As[(wm + i * 16 + ar) * LDT + ac]);
#pragma unroll
            for (int j = 0; j < 4; ++j)
                bfg[j] = ldfragB(&Bs[(wn + j * 16 + ar) * LDT + bc]);
#pragma unroll
            for (int i = 0; i < 2; ++i)
#pragma unroll
                for (int j = 0; j < 4; ++j)
                    acc[i][j] = __builtin_amdgcn_wmma_f32_16x16x32_bf16(
                        false, af[i], false, bfg[j], (short)0, acc[i][j],
                        false, false);
        }
        __syncthreads();
    }

    // ---- epilogue: C/D layout (VGPR j: lanes<16 -> M=j, lanes>=16 -> M=8+j) --
    const int col   = lane & 15;
    const int rbase = (lane < 16) ? 0 : 8;
#pragma unroll
    for (int i = 0; i < 2; ++i) {
#pragma unroll
        for (int j = 0; j < 4; ++j) {
            const long gn = n0 + wn + j * 16 + col;
            if (gn >= N) continue;
#pragma unroll
            for (int jr = 0; jr < 8; ++jr) {
                const long gm = m0 + wm + i * 16 + rbase + jr;
                if (gm >= M) continue;
                float v = acc[i][j][jr];
                if (accum) v += C[gm * ldc + gn];
                if (relu)  v = fmaxf(v, 0.f);
                C[gm * ldc + gn] = v;
            }
        }
    }
}

// ---- layernorm: one wave (32 lanes) per 256-wide row ------------------------
__global__ __launch_bounds__(256) void ln_kernel(
    const float* __restrict__ x, const float* __restrict__ w,
    const float* __restrict__ b, float* __restrict__ h, int rows)
{
    const int lane = threadIdx.x & 31, wid = threadIdx.x >> 5;
    const long row = (long)blockIdx.x * 8 + wid;
    if (row >= rows) return;
    const float* xr = x + row * RD;
    float v[8], s = 0.f, ss = 0.f;
#pragma unroll
    for (int j = 0; j < 8; ++j) {
        v[j] = xr[lane + j * 32];
        s += v[j]; ss += v[j] * v[j];
    }
#pragma unroll
    for (int m = 16; m >= 1; m >>= 1) {
        s  += __shfl_xor(s, m, 32);
        ss += __shfl_xor(ss, m, 32);
    }
    const float mean = s * (1.f / RD);
    const float var  = ss * (1.f / RD) - mean * mean;
    const float inv  = rsqrtf(var + 1e-5f);
    float* hr = h + row * RD;
#pragma unroll
    for (int j = 0; j < 8; ++j) {
        const int c = lane + j * 32;
        hr[c] = (v[j] - mean) * inv * w[c] + b[c];
    }
}

// ---- causal depthwise conv (K=4) over 7200-long sequences + SiLU ------------
__global__ __launch_bounds__(256) void dwconv_silu(
    const float* __restrict__ xz, const float* __restrict__ cw,
    const float* __restrict__ cb, float* __restrict__ xi)
{
    const long row = blockIdx.x;          // 14400 rows
    const int  d   = threadIdx.x;
    const int  tl  = (int)(row % LSEQ);
    float a = cb[d];
#pragma unroll
    for (int k = 0; k < KW; ++k) {
        const int tt = tl - (KW - 1) + k;
        if (tt >= 0) a += xz[(row - (KW - 1) + k) * (2 * DI) + d] * cw[d * KW + k];
    }
    xi[row * DI + d] = siluf(a);
}

// ---- dt = softplus(dt_l @ dtp_w + b) : K=16, one block per row --------------
__global__ __launch_bounds__(256) void dt_kernel(
    const float* __restrict__ proj, const float* __restrict__ W,
    const float* __restrict__ bias, float* __restrict__ dt)
{
    const long row = blockIdx.x;
    const int  d   = threadIdx.x;
    __shared__ float p[DTR];
    if (threadIdx.x < DTR) p[threadIdx.x] = proj[row * 48 + threadIdx.x];
    __syncthreads();
    float a = bias[d];
#pragma unroll
    for (int r = 0; r < DTR; ++r) a += p[r] * W[r * DI + d];
    dt[row * DI + d] = softplusf(a);
}

// ---- selective scan, thread = (batch, channel d), 16-state ILP --------------
__global__ __launch_bounds__(256) void scan_kernel(
    const float* __restrict__ xi, const float* __restrict__ dt,
    const float* __restrict__ proj, const float* __restrict__ xz,
    const float* __restrict__ a_log, const float* __restrict__ dv,
    float* __restrict__ y)
{
    const int b = blockIdx.x;
    const int d = threadIdx.x;
    float Ar[NSV], hs[NSV];
#pragma unroll
    for (int n = 0; n < NSV; ++n) {
        Ar[n] = -__expf(a_log[d * NSV + n]);
        hs[n] = 0.f;
    }
    const float Dv = dv[d];
    const long base = (long)b * LSEQ;
    for (int t = 0; t < LSEQ; ++t) {
        const long row = base + t;
        const float dtv = dt[row * DI + d];
        const float xv  = xi[row * DI + d];
        const float dtx = dtv * xv;
        // B (16) then C (16) live contiguously at proj row + 16; float4 loads
        const float4* pb = (const float4*)(proj + row * 48 + 16);
        float Bv[NSV], Cv[NSV];
#pragma unroll
        for (int q = 0; q < 4; ++q) {
            float4 fb = pb[q];
            Bv[4 * q + 0] = fb.x; Bv[4 * q + 1] = fb.y;
            Bv[4 * q + 2] = fb.z; Bv[4 * q + 3] = fb.w;
            float4 fc = pb[4 + q];
            Cv[4 * q + 0] = fc.x; Cv[4 * q + 1] = fc.y;
            Cv[4 * q + 2] = fc.z; Cv[4 * q + 3] = fc.w;
        }
        float yv = 0.f;
#pragma unroll
        for (int n = 0; n < NSV; ++n) {
            hs[n] = __expf(dtv * Ar[n]) * hs[n] + dtx * Bv[n];
            yv   += hs[n] * Cv[n];
        }
        const float zv = xz[row * (2 * DI) + DI + d];
        y[row * DI + d] = (yv + xv * Dv) * siluf(zv);
    }
}

// ---- final unmerge: x[b][t][d] -> (F_q, F_s)[b][d][l] -----------------------
__global__ __launch_bounds__(256) void unmerge(
    const float* __restrict__ x, float* __restrict__ out)
{
    const long idx = (long)blockIdx.x * 256 + threadIdx.x;
    const long TS = (long)BATCH * RD * LSP;
    const int  q  = (int)(idx / TS);
    long r        = idx % TS;
    const int  b  = (int)(r / ((long)RD * LSP));
    r             = r % ((long)RD * LSP);
    const int  dd = (int)(r / LSP);
    const int  l  = (int)(r % LSP);
    out[idx] = x[((long)b * LSEQ + (long)q * LSP + l) * RD + dd];
}

extern "C" void kernel_launch(void* const* d_in, const int* in_sizes, int n_in,
                              void* d_out, int out_size, void* d_ws, size_t ws_size,
                              hipStream_t stream)
{
    (void)in_sizes; (void)n_in; (void)out_size; (void)ws_size;
    const float* F_q       = (const float*)d_in[0];
    const float* F_s       = (const float*)d_in[1];
    const float* down_q_w  = (const float*)d_in[2];
    const float* down_s_w  = (const float*)d_in[3];
    const float* merge_q_w = (const float*)d_in[4];
    const float* merge_s_w = (const float*)d_in[5];
    const float* ln_w      = (const float*)d_in[6];
    const float* ln_b      = (const float*)d_in[7];
    const float* in_proj_w = (const float*)d_in[8];
    const float* conv_w    = (const float*)d_in[9];
    const float* conv_b    = (const float*)d_in[10];
    const float* x_proj_w  = (const float*)d_in[11];
    const float* dt_proj_w = (const float*)d_in[12];
    const float* dt_proj_b = (const float*)d_in[13];
    const float* A_log     = (const float*)d_in[14];
    const float* D_p       = (const float*)d_in[15];
    const float* out_proj_w= (const float*)d_in[16];
    float* out = (float*)d_out;

    char* base = (char*)d_ws;
    size_t off = 0;
    auto alloc = [&](size_t bytes) -> char* {
        off = (off + 255) & ~(size_t)255;
        char* p = base + off;
        off += bytes;
        return p;
    };
    // bf16 weights (B^T layout)
    bf16_t* wdq   = (bf16_t*)alloc((size_t)RD * 1536 * 2);
    bf16_t* wds   = (bf16_t*)alloc((size_t)RD * 1536 * 2);
    bf16_t* wmq   = (bf16_t*)alloc((size_t)RD * 512 * 2);
    bf16_t* wms   = (bf16_t*)alloc((size_t)RD * 512 * 2);
    bf16_t* winp  = (bf16_t*)alloc((size_t)DEPTH * 512 * RD * 2);
    bf16_t* wxp   = (bf16_t*)alloc((size_t)DEPTH * 48 * DI * 2);
    bf16_t* woutp = (bf16_t*)alloc((size_t)DEPTH * RD * DI * 2);
    // f32 activations (xz doubles as fqs; h doubles as y)
    float* xz   = (float*)alloc((size_t)ROWS * 512 * 4);
    float* xbuf = (float*)alloc((size_t)ROWS * RD * 4);
    float* h    = (float*)alloc((size_t)ROWS * RD * 4);
    float* xi   = (float*)alloc((size_t)ROWS * DI * 4);
    float* proj = (float*)alloc((size_t)ROWS * 48 * 4);
    float* dtb  = (float*)alloc((size_t)ROWS * DI * 4);
    float* fqs  = xz;
    float* y    = h;

    auto pack = [&](const float* src, bf16_t* dst, int N, int K, long sn, long sk,
                    int rot, long lss, long lsd, int layers) {
        dim3 g((unsigned)(((long)N * K + 255) / 256), 1, (unsigned)layers);
        pack_bt<<<g, 256, 0, stream>>>(src, dst, N, K, sn, sk, rot, lss, lsd);
    };
    auto gemm = [&](const float* A, const bf16_t* BT, float* C, int M, int N, int K,
                    long sAm, long sAk, int ldc, int relu, int accum) {
        dim3 g((unsigned)((M + BM - 1) / BM), (unsigned)((N + BN - 1) / BN));
        gemm_bt<<<g, 256, 0, stream>>>(A, BT, C, M, N, K, sAm, sAk, ldc, relu, accum);
    };

    // 1) repack weights to bf16 B^T
    pack(down_q_w,  wdq,  RD, 1536, 1536, 1, 0,   0, 0, 1);
    pack(down_s_w,  wds,  RD, 1536, 1536, 1, 0,   0, 0, 1);
    pack(merge_q_w, wmq,  RD, 512,  512,  1, 0,   0, 0, 1);
    pack(merge_s_w, wms,  RD, 512,  512,  1, 256, 0, 0, 1);  // swap fs/fq halves
    pack(in_proj_w, winp, 512, RD,  1, 512, 0, (long)RD * 512, (long)512 * RD, DEPTH);
    pack(x_proj_w,  wxp,  48,  DI,  1, 48,  0, (long)DI * 48,  (long)48 * DI,  DEPTH);
    pack(out_proj_w,woutp, RD, DI,  1, RD,  0, (long)DI * RD,  (long)RD * DI,  DEPTH);

    // 2) down projections (A = F[b] viewed as [L x C], strided) + ReLU
    for (int b = 0; b < BATCH; ++b) {
        gemm(F_q + (long)b * 1536 * LSP, wdq, fqs + (long)b * LSP * 512,
             LSP, RD, 1536, 1, LSP, 512, 1, 0);
        gemm(F_s + (long)b * 1536 * LSP, wds, fqs + (long)b * LSP * 512 + RD,
             LSP, RD, 1536, 1, LSP, 512, 1, 0);
    }
    // 3) merge projections -> x (q rows then s rows per batch) + ReLU
    for (int b = 0; b < BATCH; ++b) {
        gemm(fqs + (long)b * LSP * 512, wmq, xbuf + ((long)b * LSEQ) * RD,
             LSP, RD, 512, 512, 1, RD, 1, 0);
        gemm(fqs + (long)b * LSP * 512, wms, xbuf + ((long)b * LSEQ + LSP) * RD,
             LSP, RD, 512, 512, 1, RD, 1, 0);
    }
    // 4) mamba layers
    for (int i = 0; i < DEPTH; ++i) {
        ln_kernel<<<ROWS / 8, 256, 0, stream>>>(xbuf, ln_w + (long)i * RD,
                                                ln_b + (long)i * RD, h, ROWS);
        gemm(h, winp + (long)i * 512 * RD, xz, ROWS, 512, RD, RD, 1, 512, 0, 0);
        dwconv_silu<<<ROWS, 256, 0, stream>>>(xz, conv_w + (long)i * DI * KW,
                                              conv_b + (long)i * DI, xi);
        gemm(xi, wxp + (long)i * 48 * DI, proj, ROWS, 48, DI, DI, 1, 48, 0, 0);
        dt_kernel<<<ROWS, 256, 0, stream>>>(proj, dt_proj_w + (long)i * DTR * DI,
                                            dt_proj_b + (long)i * DI, dtb);
        scan_kernel<<<BATCH, 256, 0, stream>>>(xi, dtb, proj, xz,
                                               A_log + (long)i * DI * NSV,
                                               D_p + (long)i * DI, y);
        gemm(y, woutp + (long)i * RD * DI, xbuf, ROWS, RD, DI, DI, 1, RD, 0, 1);
    }
    // 5) unmerge to output layout
    unmerge<<<(unsigned)(((long)2 * BATCH * RD * LSP) / 256), 256, 0, stream>>>(xbuf, out);
}
